// GraphAttentionLayer_22308060136177
// MI455X (gfx1250) — compile-verified, matrix operand order checked by reading, not run on hardware
//
#include <hip/hip_runtime.h>
#include <hip/hip_bf16.h>

// ---------------------------------------------------------------------------
// GAT layer for MI455X (gfx1250, wave32).
//   GEMM1 (z = h*W^T + b)        : v_wmma_f32_16x16x4_f32  (fp32 precision)
//   GEMM2 (h' = softmax(e) * z)  : v_wmma_f32_16x16x32_f16 (f16 in, f32 acc)
//   Bulk tile movement            : TENSOR_LOAD_TO_LDS (TDM), double-buffered
// ---------------------------------------------------------------------------

typedef __attribute__((ext_vector_type(2)))  float        v2f;
typedef __attribute__((ext_vector_type(8)))  float        v8f;
typedef __attribute__((ext_vector_type(8)))  _Float16     v8h;
typedef __attribute__((ext_vector_type(16))) _Float16     v16h;
typedef __attribute__((ext_vector_type(4)))  unsigned int v4u;
typedef __attribute__((ext_vector_type(8)))  int          v8i;
typedef __attribute__((ext_vector_type(4)))  int          v4i;

#define GAT_B   32
#define GAT_N   1024
#define GAT_D   256
#define GAT_R   (GAT_B * GAT_N)
#define ALPHA_LRELU 0.2f
#define NEG_MASK    (-9000000000000000.0f)

union U16H { v16h v; v8h h[2]; };

// --- LDS byte offset of a __shared__ object: flat-LDS addresses map to the
//     LDS offset in the low 32 bits (ISA 10.2: upper bits discarded). --------
__device__ __forceinline__ unsigned int lds_off(const void* p) {
    return (unsigned int)(uintptr_t)p;
}

// --- TDM: build D# (groups 0/1 per ISA 08_async_tensor 8.3/8.4) and issue
//     TENSOR_LOAD_TO_LDS (6-arg toolchain variant). 2-D tile (tile_y rows of
//     tile_x elems, row stride `stride_elems`). data_size_code: 2 = 4B.
//     w0_extra carries pad_enable/pad_interval/pad_amount bits. -------------
__device__ __forceinline__ void tdm_load_tile(unsigned int lds_byte,
                                              const void* gptr,
                                              unsigned int tile_x,
                                              unsigned int tile_y,
                                              unsigned long long stride_elems,
                                              unsigned int data_size_code,
                                              unsigned int w0_extra) {
    unsigned long long ga = (unsigned long long)(uintptr_t)gptr;
    v4u g0;
    g0.x = 1u;                                            // count=1, user mode
    g0.y = lds_byte;                                      // lds_addr
    g0.z = (unsigned int)ga;                              // global_addr[31:0]
    g0.w = (unsigned int)((ga >> 32) & 0x01FFFFFFu)       // global_addr[56:32]
         | (2u << 30);                                    // type = 2 ("image")
    const unsigned int td0 = 0x3FFFFFFFu, td1 = 0x3FFFFFFFu; // dims: no OOB clip
    v8i g1;
    g1[0] = (int)((data_size_code << 16) | w0_extra);     // wg_mask=0, flags
    g1[1] = (int)((td0 & 0xFFFFu) << 16);                 // tensor_dim0 lo
    g1[2] = (int)(((td0 >> 16) & 0xFFFFu) | ((td1 & 0xFFFFu) << 16));
    g1[3] = (int)(((td1 >> 16) & 0xFFFFu) | (tile_x << 16));
    g1[4] = (int)(tile_y & 0xFFFFu);                      // tile_dim1, tile_dim2=0
    g1[5] = (int)(unsigned int)(stride_elems & 0xFFFFFFFFu);
    g1[6] = (int)(unsigned int)((stride_elems >> 32) & 0xFFFFu);
    g1[7] = 0;
    v4i gz4 = {};
    v8i gz8 = {};
    __builtin_amdgcn_tensor_load_to_lds(g0, g1, gz4, gz4, gz8, 0);
}

// --- ds_swizzle XOR-butterfly sum over the 16-lane half (masks 1,2,4,8) ----
__device__ __forceinline__ float half16_xor_sum(float x) {
    x += __int_as_float(__builtin_amdgcn_ds_swizzle(__float_as_int(x), 0x1f | (1 << 10)));
    x += __int_as_float(__builtin_amdgcn_ds_swizzle(__float_as_int(x), 0x1f | (2 << 10)));
    x += __int_as_float(__builtin_amdgcn_ds_swizzle(__float_as_int(x), 0x1f | (4 << 10)));
    x += __int_as_float(__builtin_amdgcn_ds_swizzle(__float_as_int(x), 0x1f | (8 << 10)));
    return x;
}

__device__ __forceinline__ float lrelu(float x) { return x > 0.f ? x : ALPHA_LRELU * x; }

// ---------------------------------------------------------------------------
// Kernel 1: z tile (16x256) via f32 WMMA; emit zT (f16), s1, s2.
// h tile staged by one TDM op; D# padding gives the 260-float bank-safe rows.
// B-fragments read W untransposed: lane K-pair (kk, kk+1) is one b64 load.
// ---------------------------------------------------------------------------
__global__ void __launch_bounds__(256)
k1_proj(const float* __restrict__ h, const float* __restrict__ W,
        const float* __restrict__ bias, const float* __restrict__ avec,
        _Float16* __restrict__ zT, float* __restrict__ s1, float* __restrict__ s2) {
    constexpr int HS = 260;                 // 256 + TDM pad (4 DW per 256 DW)
    __shared__ float hs[16 * HS];
    __shared__ float l_s1[16], l_s2[16];

    const int tid  = threadIdx.x;
    const int lane = tid & 31, wave = tid >> 5;
    const int n    = lane & 15;
    const int half = lane >> 4;

    const int r0 = blockIdx.x * 16;
    const int b  = r0 >> 10;
    const int i0 = r0 & (GAT_N - 1);

    if (wave == 0) {
        // 1-D tile of 4096 f32, pad_enable, pad_interval=7 (256 DW),
        // pad_amount=3 (4 DW)  ->  row stride 260 floats in LDS.
        tdm_load_tile(lds_off(hs), h + (size_t)r0 * GAT_D,
                      /*tile_x=*/4096, /*tile_y=*/0, /*stride=*/0,
                      /*dsz=*/2u,
                      /*w0=*/(1u << 20) | (7u << 22) | (3u << 25));
        __builtin_amdgcn_s_wait_tensorcnt(0);
    }
    if (tid < 16) { l_s1[tid] = 0.f; l_s2[tid] = 0.f; }
    __syncthreads();

    const int col0 = wave * 32;
    v8f acc[2] = {v8f{0.f}, v8f{0.f}};

    for (int k = 0; k < GAT_D; k += 4) {
        const int kk = k + half * 2;                 // f32 A/B: K = half*2 + v
        const float2 a2 = *(const float2*)(&hs[n * HS + kk]);
        v2f af; af.x = a2.x; af.y = a2.y;
        #pragma unroll
        for (int ct = 0; ct < 2; ++ct) {
            const int c = col0 + ct * 16 + n;
            // B[K=kk+v][N=c] = W[c*256 + kk+v] : contiguous pair -> one b64
            const float2 w2 = *(const float2*)(&W[(size_t)c * GAT_D + kk]);
            v2f bf; bf.x = w2.x; bf.y = w2.y;
            acc[ct] = __builtin_amdgcn_wmma_f32_16x16x4_f32(
                false, af, false, bf, (short)0, acc[ct], false, false);
        }
    }

    #pragma unroll
    for (int ct = 0; ct < 2; ++ct) {
        const int c   = col0 + ct * 16 + n;
        const float bv  = bias[c];
        const float a1v = avec[c];
        const float a2v = avec[GAT_D + c];
        v8h zrow;
        #pragma unroll
        for (int v = 0; v < 8; ++v) {
            float cv = acc[ct][v] + bv;
            zrow[v] = (_Float16)cv;
            float t1 = half16_xor_sum(cv * a1v);
            float t2 = half16_xor_sum(cv * a2v);
            if (n == 0) {
                atomicAdd(&l_s1[v + half * 8], t1);
                atomicAdd(&l_s2[v + half * 8], t2);
            }
        }
        *(v8h*)(zT + ((size_t)(b * GAT_D + c)) * GAT_N + i0 + half * 8) = zrow;
    }
    __syncthreads();
    if (tid < 16) { s1[r0 + tid] = l_s1[tid]; s2[r0 + tid] = l_s2[tid]; }
}

// ---------------------------------------------------------------------------
// Kernel 2: per-row softmax stats over masked rank-1 logits.
// ---------------------------------------------------------------------------
__global__ void __launch_bounds__(256)
k2_stats(const int* __restrict__ adj, const float* __restrict__ s1,
         const float* __restrict__ s2, float* __restrict__ mrow,
         float* __restrict__ izrow) {
    __shared__ float ev[GAT_N];
    __shared__ float red[8];
    __shared__ float bcast;

    const int tid = threadIdx.x, lane = tid & 31, wave = tid >> 5;
    const int row = blockIdx.x;
    const int b   = row >> 10;
    const float s1i = s1[row];

    float lmax = -3.4e38f;
    for (int j = tid; j < GAT_N; j += 256) {
        int   aij = adj[(size_t)row * GAT_N + j];
        float e   = (aij > 0) ? lrelu(s1i + s2[(b << 10) + j]) : NEG_MASK;
        ev[j] = e;
        lmax  = fmaxf(lmax, e);
    }
    #pragma unroll
    for (int off = 16; off > 0; off >>= 1) lmax = fmaxf(lmax, __shfl_xor(lmax, off, 32));
    if (lane == 0) red[wave] = lmax;
    __syncthreads();
    if (tid == 0) {
        float m = red[0];
        for (int w = 1; w < 8; ++w) m = fmaxf(m, red[w]);
        bcast = m;
    }
    __syncthreads();
    const float m = bcast;

    float lsum = 0.f;
    for (int j = tid; j < GAT_N; j += 256) lsum += __expf(ev[j] - m);
    #pragma unroll
    for (int off = 16; off > 0; off >>= 1) lsum += __shfl_xor(lsum, off, 32);
    if (lane == 0) red[wave] = lsum;
    __syncthreads();
    if (tid == 0) {
        float z = 0.f;
        for (int w = 0; w < 8; ++w) z += red[w];
        mrow[row]  = m;
        izrow[row] = 1.0f / z;
    }
}

// ---------------------------------------------------------------------------
// Kernel 3: h' = A @ z. adj tiles arrive via double-buffered TDM (fetch of
// chunk c+1 overlaps the WMMA sweep of chunk c); A tile built in f16 LDS;
// f16 WMMA against zT; fused ELU.
// ---------------------------------------------------------------------------
__global__ void __launch_bounds__(256)
k3_attn(const int* __restrict__ adj, const float* __restrict__ s1,
        const float* __restrict__ s2, const float* __restrict__ mrow,
        const float* __restrict__ izrow, const _Float16* __restrict__ zT,
        float* __restrict__ out) {
    constexpr int AS = 264;
    __shared__ int      adjS[2][16 * 256];           // 32 KB double buffer
    __shared__ _Float16 At[16 * AS];                 // 8.25 KB A tile
    __shared__ float    s2sh[GAT_N];
    __shared__ float    s1sh[16], msh[16], izsh[16];

    const int tid  = threadIdx.x;
    const int lane = tid & 31, wave = tid >> 5;
    const int n    = lane & 15;
    const int half = lane >> 4;
    const int ksel = half * 8;

    const int r0 = blockIdx.x * 16;
    const int b  = r0 >> 10;
    const int i0 = r0 & (GAT_N - 1);
    const size_t arow0 = (size_t)(b * GAT_N + i0) * GAT_N;

    if (wave == 0)                                   // prefetch chunk 0
        tdm_load_tile(lds_off(&adjS[0][0]), adj + arow0,
                      /*tile_x=*/256, /*tile_y=*/16, /*stride=*/GAT_N,
                      /*dsz=*/2u, /*w0=*/0u);

    for (int j = tid; j < GAT_N; j += 256) s2sh[j] = s2[(b << 10) + j];
    if (tid < 16) {
        s1sh[tid] = s1[r0 + tid];
        msh[tid]  = mrow[r0 + tid];
        izsh[tid] = izrow[r0 + tid];
    }

    v8f acc[2] = {v8f{0.f}, v8f{0.f}};
    const int col0 = wave * 32;

    for (int c = 0; c < 4; ++c) {
        const int jc  = c * 256;
        const int buf = c & 1;
        if (wave == 0) __builtin_amdgcn_s_wait_tensorcnt(0);
        __syncthreads();                             // adjS[buf] ready; Atile free
        if (wave == 0 && c + 1 < 4)                  // overlap next DMA with math
            tdm_load_tile(lds_off(&adjS[(c + 1) & 1][0]),
                          adj + arow0 + jc + 256,
                          256, 16, GAT_N, 2u, 0u);
        #pragma unroll
        for (int r = 0; r < 16; ++r) {
            int aij = adjS[buf][r * 256 + tid];
            float e = (aij > 0) ? lrelu(s1sh[r] + s2sh[jc + tid]) : NEG_MASK;
            At[r * AS + tid] = (_Float16)(__expf(e - msh[r]) * izsh[r]);
        }
        __syncthreads();

        for (int jj = 0; jj < 256; jj += 32) {
            U16H af;
            af.h[0] = *(const v8h*)(&At[n * AS + jj + ksel]);
            af.h[1] = *(const v8h*)(&At[n * AS + jj + ksel + 16]);
            #pragma unroll
            for (int ct = 0; ct < 2; ++ct) {
                const int c2 = col0 + ct * 16 + n;
                const size_t base = ((size_t)(b * GAT_D + c2)) * GAT_N + jc + jj + ksel;
                U16H bf;
                bf.h[0] = *(const v8h*)(zT + base);
                bf.h[1] = *(const v8h*)(zT + base + 16);
                acc[ct] = __builtin_amdgcn_wmma_f32_16x16x32_f16(
                    false, af.v, false, bf.v, (short)0, acc[ct], false, false);
            }
        }
    }

    #pragma unroll
    for (int ct = 0; ct < 2; ++ct) {
        const int c2 = col0 + ct * 16 + n;
        #pragma unroll
        for (int v = 0; v < 8; ++v) {
            float x = acc[ct][v];
            float y = (x > 0.f) ? x : (__expf(x) - 1.0f);
            out[(size_t)(b * GAT_N + i0 + half * 8 + v) * GAT_D + c2] = y;
        }
    }
}

// ---------------------------------------------------------------------------
// Launch
// ---------------------------------------------------------------------------
extern "C" void kernel_launch(void* const* d_in, const int* in_sizes, int n_in,
                              void* d_out, int out_size, void* d_ws, size_t ws_size,
                              hipStream_t stream) {
    const float* h    = (const float*)d_in[0];
    const int*   adj  = (const int*)  d_in[1];
    const float* W    = (const float*)d_in[2];
    const float* bias = (const float*)d_in[3];
    const float* avec = (const float*)d_in[4];
    float*       out  = (float*)d_out;

    char* ws = (char*)d_ws;
    _Float16* zT = (_Float16*)ws;
    size_t off = (size_t)GAT_B * GAT_D * GAT_N * sizeof(_Float16);
    float* s1 = (float*)(ws + off);   off += (size_t)GAT_R * sizeof(float);
    float* s2 = (float*)(ws + off);   off += (size_t)GAT_R * sizeof(float);
    float* mr = (float*)(ws + off);   off += (size_t)GAT_R * sizeof(float);
    float* iz = (float*)(ws + off);

    k1_proj<<<GAT_R / 16, 256, 0, stream>>>(h, W, bias, avec, zT, s1, s2);
    k2_stats<<<GAT_R, 256, 0, stream>>>(adj, s1, s2, mr, iz);
    k3_attn<<<GAT_R / 16, 256, 0, stream>>>(adj, s1, s2, mr, iz, zT, out);
}
